// GCN_69956427317969
// MI455X (gfx1250) — matrix operand order repaired and checked by reading, not compiled, not run on hardware
//
#include <hip/hip_runtime.h>
#include <hip/hip_bf16.h>
#include <math.h>

typedef __attribute__((ext_vector_type(16))) __bf16 v16bf;
typedef __attribute__((ext_vector_type(8)))  __bf16 v8bf;
typedef __attribute__((ext_vector_type(4)))  __bf16 v4bf;
typedef __attribute__((ext_vector_type(8)))  float  v8f;

#define N_NODES   100000
#define N_EDGES   640000
#define F_INDIM   128
#define N_CLASSES 64

// ---------------- degree / norm ----------------
__global__ void deg_init_kernel(float* __restrict__ deg, int n) {
    int i = blockIdx.x * blockDim.x + threadIdx.x;
    if (i < n) deg[i] = 1.0f;                 // self-loop contributes 1
}

__global__ void deg_count_kernel(const int* __restrict__ dst,
                                 float* __restrict__ deg, int e) {
    int i = blockIdx.x * blockDim.x + threadIdx.x;
    if (i < e) atomicAdd(&deg[dst[i]], 1.0f);
}

__global__ void dinv_kernel(float* __restrict__ deg, int n) {
    int i = blockIdx.x * blockDim.x + threadIdx.x;
    if (i < n) deg[i] = rsqrtf(deg[i]);       // deg >= 1 always
}

// ---------------- W pre-transpose+convert: Wt[n*128+k] = bf16(W[k*FOUT+n]) ---
template<int FOUT>
__global__ void transpose_w_bf16(const float* __restrict__ W,
                                 __bf16* __restrict__ Wt) {
    int i = blockIdx.x * blockDim.x + threadIdx.x;     // over FOUT*128
    if (i >= FOUT * F_INDIM) return;
    int n = i >> 7;            // output column
    int k = i & 127;           // input feature
    Wt[i] = (__bf16)W[k * FOUT + n];
}

// ---------------- GEMM: H[16xFOUT] per block = Xtile[16x128] * W[128xFOUT] ---
// A tile staged once per block into LDS as bf16 (row-major, 4KB).
// Per K-step: A frag = 2x ds_load_b128 (two contiguous 8-elem runs),
//             B frag = 2x 16B loads (two 8-elem runs) from bf16 transposed Wt (L2-resident).
template<int FOUT>
__global__ void gemm_wmma_bf16(const float* __restrict__ X,
                               const __bf16* __restrict__ Wt,
                               float* __restrict__ H) {
    __shared__ __bf16 tileA[16 * F_INDIM];             // 4 KB
    const int tid  = threadIdx.x;
    const int rowB = blockIdx.x * 16;

    // ---- cooperative stage: 16x128 tile of X is a contiguous 8KB span ----
    const float4* Xv = (const float4*)X;
    for (int i = tid; i < 512; i += blockDim.x) {      // 512 float4s
        float4 f = Xv[(size_t)rowB * 32 + i];
        v4bf o;
        o[0] = (__bf16)f.x; o[1] = (__bf16)f.y;
        o[2] = (__bf16)f.z; o[3] = (__bf16)f.w;
        *(v4bf*)&tileA[i * 4] = o;
    }
    __syncthreads();

    const int lane  = tid & 31;
    const int ctile = tid >> 5;
    const int half  = lane >> 4;
    const int l16   = lane & 15;
    const int ncol  = ctile * 16 + l16;

    v8f acc = {};
#pragma unroll
    for (int kb = 0; kb < F_INDIM / 32; ++kb) {
        // A 16x32 bf16 fragment (ISA 7.12.2): elems 0..7  = K=kb*32+half*8+{0..7}
        //                                     elems 8..15 = +16
        int abase = l16 * F_INDIM + kb * 32 + half * 8;
        v8bf alo = *(const v8bf*)&tileA[abase];
        v8bf ahi = *(const v8bf*)&tileA[abase + 16];
        v16bf a = __builtin_shufflevector(alo, ahi,
                      0, 1, 2, 3, 4, 5, 6, 7, 8, 9, 10, 11, 12, 13, 14, 15);
        // B 32x16 fragment: same K layout as A (two 8-elem runs) in transposed Wt
        int bbase = ncol * F_INDIM + kb * 32 + half * 8;
        v8bf blo = *(const v8bf*)&Wt[bbase];
        v8bf bhi = *(const v8bf*)&Wt[bbase + 16];
        v16bf b = __builtin_shufflevector(blo, bhi,
                      0, 1, 2, 3, 4, 5, 6, 7, 8, 9, 10, 11, 12, 13, 14, 15);
        acc = __builtin_amdgcn_wmma_f32_16x16x32_bf16(
                  false, a, false, b, (short)0, acc, false, false);
    }
#pragma unroll
    for (int r = 0; r < 8; ++r) {
        int m = r + (half << 3);            // C/D: VGPR r -> M=r (+8 for hi half)
        H[(rowB + m) * FOUT + ncol] = acc[r];
    }
}

// ---------------- aggregation ----------------
// agg[i,f] = dinv[i]^2 * h[i,f]   (self-loop term; also initializes agg)
template<int F>
__global__ void self_loop_kernel(const float* __restrict__ dinv,
                                 const float* __restrict__ h,
                                 float* __restrict__ agg, int total) {
    int i = blockIdx.x * blockDim.x + threadIdx.x;
    if (i >= total) return;
    int node = i / F;
    float w = dinv[node];
    agg[i] = w * w * h[i];
}

// one wave per edge; vectorized gather, per-float atomics to dst row
template<int F>
__global__ void scatter_kernel(const int* __restrict__ src,
                               const int* __restrict__ dst,
                               const float* __restrict__ dinv,
                               const float* __restrict__ h,
                               float* __restrict__ agg, int nE) {
    int gid  = blockIdx.x * blockDim.x + threadIdx.x;
    int e    = gid >> 5;
    int lane = gid & 31;
    if (e >= nE) return;
    int s = src[e];
    int d = dst[e];
    float w = dinv[s] * dinv[d];
    if (F >= 128) {           // 4 features per lane: one b128 gather
        const float4* hv = (const float4*)(h + (size_t)s * F);
        float4 v = hv[lane];
        float* a = agg + (size_t)d * F + lane * 4;
        atomicAdd(a + 0, w * v.x);
        atomicAdd(a + 1, w * v.y);
        atomicAdd(a + 2, w * v.z);
        atomicAdd(a + 3, w * v.w);
    } else {                  // F==64: 2 features per lane
        const float2* hv = (const float2*)(h + (size_t)s * F);
        float2 v = hv[lane];
        float* a = agg + (size_t)d * F + lane * 2;
        atomicAdd(a + 0, w * v.x);
        atomicAdd(a + 1, w * v.y);
    }
}

template<int F, bool RELU>
__global__ void bias_act_kernel(float* __restrict__ agg,
                                const float* __restrict__ bias, int total) {
    int i = blockIdx.x * blockDim.x + threadIdx.x;
    if (i >= total) return;
    float v = agg[i] + bias[i % F];
    agg[i] = RELU ? fmaxf(v, 0.0f) : v;
}

// ---------------- log_softmax over 64 classes, bias fused, wave per node ----
__global__ void log_softmax_kernel(const float* __restrict__ z,
                                   const float* __restrict__ bias,
                                   float* __restrict__ out, int n) {
    int gid  = blockIdx.x * blockDim.x + threadIdx.x;
    int node = gid >> 5;
    int lane = gid & 31;
    if (node >= n) return;
    float z0 = z[node * N_CLASSES + lane]      + bias[lane];
    float z1 = z[node * N_CLASSES + 32 + lane] + bias[32 + lane];
    float m = fmaxf(z0, z1);
#pragma unroll
    for (int off = 16; off > 0; off >>= 1)
        m = fmaxf(m, __shfl_xor(m, off, 32));
    float s = expf(z0 - m) + expf(z1 - m);
#pragma unroll
    for (int off = 16; off > 0; off >>= 1)
        s += __shfl_xor(s, off, 32);
    float lse = logf(s) + m;
    out[node * N_CLASSES + lane]      = z0 - lse;
    out[node * N_CLASSES + 32 + lane] = z1 - lse;
}

// ---------------- launch ----------------
extern "C" void kernel_launch(void* const* d_in, const int* in_sizes, int n_in,
                              void* d_out, int out_size, void* d_ws, size_t ws_size,
                              hipStream_t stream) {
    const float* x  = (const float*)d_in[0];
    const int*   eg = (const int*)d_in[1];
    const int*   src = eg;              // edge[0]
    const int*   dst = eg + N_EDGES;    // edge[1]
    const float* W1 = (const float*)d_in[2];
    const float* b1 = (const float*)d_in[3];
    const float* W2 = (const float*)d_in[4];
    const float* b2 = (const float*)d_in[5];
    float* out = (float*)d_out;

    char* ws = (char*)d_ws;
    size_t featBytes = (size_t)N_NODES * F_INDIM * sizeof(float);    // 51.2 MB
    float*  h    = (float*)ws;                         // h1, reused as h2
    float*  agg  = (float*)(ws + featBytes);           // agg1/out1, reused as agg2
    float*  dinv = (float*)(ws + 2 * featBytes);       // degrees -> dinv (100k f32)
    __bf16* Wt1  = (__bf16*)(ws + 2 * featBytes + (512ull << 10));           // 32 KB
    __bf16* Wt2  = (__bf16*)(ws + 2 * featBytes + (512ull << 10) + (64 << 10)); // 16 KB

    const int B = 256;

    // degrees & symmetric norm; W conversions (independent, same stream)
    deg_init_kernel<<<(N_NODES + B - 1) / B, B, 0, stream>>>(dinv, N_NODES);
    deg_count_kernel<<<(N_EDGES + B - 1) / B, B, 0, stream>>>(dst, dinv, N_EDGES);
    dinv_kernel<<<(N_NODES + B - 1) / B, B, 0, stream>>>(dinv, N_NODES);
    transpose_w_bf16<F_INDIM><<<(F_INDIM * F_INDIM + B - 1) / B, B, 0, stream>>>(W1, Wt1);
    transpose_w_bf16<N_CLASSES><<<(N_CLASSES * F_INDIM + B - 1) / B, B, 0, stream>>>(W2, Wt2);

    // ---- layer 1: h = x @ W1 ; agg = D^-1/2 (A+I) D^-1/2 h ; relu(agg+b1) ----
    gemm_wmma_bf16<F_INDIM><<<N_NODES / 16, 32 * (F_INDIM / 16), 0, stream>>>(x, Wt1, h);
    {
        int total = N_NODES * F_INDIM;
        self_loop_kernel<F_INDIM><<<(total + B - 1) / B, B, 0, stream>>>(dinv, h, agg, total);
        scatter_kernel<F_INDIM><<<((size_t)N_EDGES * 32 + B - 1) / B, B, 0, stream>>>(
            src, dst, dinv, h, agg, N_EDGES);
        bias_act_kernel<F_INDIM, true><<<(total + B - 1) / B, B, 0, stream>>>(agg, b1, total);
    }

    // ---- layer 2: h2 = out1 @ W2 ; agg2 ; log_softmax(agg2 + b2) ----
    gemm_wmma_bf16<N_CLASSES><<<N_NODES / 16, 32 * (N_CLASSES / 16), 0, stream>>>(agg, Wt2, h);
    {
        int total = N_NODES * N_CLASSES;
        self_loop_kernel<N_CLASSES><<<(total + B - 1) / B, B, 0, stream>>>(dinv, h, agg, total);
        scatter_kernel<N_CLASSES><<<((size_t)N_EDGES * 32 + B - 1) / B, B, 0, stream>>>(
            src, dst, dinv, h, agg, N_EDGES);
        log_softmax_kernel<<<((size_t)N_NODES * 32 + B - 1) / B, B, 0, stream>>>(
            agg, b2, out, N_NODES);
    }
}
